// SpatialEmbLoss_55843164782648
// MI455X (gfx1250) — matrix operand order repaired and checked by reading, not compiled, not run on hardware
//
#include <hip/hip_runtime.h>
#include <math.h>

// ---------------------------------------------------------------------------
// SpatialEmbLoss for MI455X (gfx1250).
// B=4, C=4, H=512, W=1024, K_MAX=8. Output: single f32 scalar.
//
// Pipeline:
//   k_zero      : zero the small header region of d_ws (stats/params/lossAcc)
//   k_stats     : WMMA (v_wmma_f32_16x16x32_f16) masked-reduction GEMM:
//                 D[k,c] = sum_p mask_k(p) * val_c(p), inner dim = 524288
//   k_finalize  : centers, s_mean, var, s_exp, present, objf; adds var+seed_bg
//   per image (segments batched by ws size):
//     k_genkeys : dist/errors -> descending-orderable u32 key, gt in LSB;
//                 accumulates seed_fg
//     4x radix pass: k_hist -> k_scanhist -> k_scatter
//                    (scatter tiles loaded via global_load_async_to_lds_b128)
//     k_gtsum / k_gtscan : segmented scan of gt bits (block level)
//     k_lovasz  : jaccard grads + weighted reduction into lossAcc[b]
//   k_final     : mean over images -> d_out[0]
// ---------------------------------------------------------------------------

#define HWPX   (512 * 1024)
#define IMG_W  1024
#define KMAX   8
#define NBLK   128      // HWPX / TILE
#define TILE   4096     // 256 threads * 16 elems
#define W_INST 1.0f
#define W_VAR  10.0f
#define W_SEED 1.0f
#define FG_W   1.0f

typedef __attribute__((ext_vector_type(16))) _Float16 v16h;
typedef __attribute__((ext_vector_type(8)))  float    v8f;

__device__ __forceinline__ float fast_sigmoid(float x) {
  return __builtin_amdgcn_rcpf(1.0f + expf(-x));   // v_rcp_f32, ~1 ulp
}

// ---------------- wave32 shuffle-based inclusive scan ----------------------
__device__ __forceinline__ unsigned waveScanIncl(unsigned v, int lane) {
  #pragma unroll
  for (int off = 1; off < 32; off <<= 1) {
    const unsigned n = __shfl_up(v, off, 32);
    if (lane >= off) v += n;
  }
  return v;
}

// inclusive scan over 256 threads; s needs >= 8 entries of scratch.
// thread 255's return value is the block total.
__device__ __forceinline__ unsigned scanBlockIncl(unsigned v, unsigned* s, int t) {
  const int lane = t & 31, wv = t >> 5;
  __syncthreads();                       // protect previous users of s
  const unsigned wi = waveScanIncl(v, lane);
  if (lane == 31) s[wv] = wi;
  __syncthreads();
  if (wv == 0) {
    unsigned x  = (lane < 8) ? s[lane] : 0u;
    unsigned xs = waveScanIncl(x, lane);
    if (lane < 8) s[lane] = xs - x;      // exclusive wave offsets
  }
  __syncthreads();
  return wi + s[wv];
}

__device__ __forceinline__ float blockReduce256(float v, float* red, int t) {
  red[t] = v;
  __syncthreads();
  for (int off = 128; off > 0; off >>= 1) {
    if (t < off) red[t] += red[t + off];
    __syncthreads();
  }
  return red[0];
}

// ---------------- header zeroing -------------------------------------------
__global__ void k_zero(float* hdr) {
  for (int i = threadIdx.x; i < 1024; i += 256) hdr[i] = 0.0f;
}

// ---------------- WMMA masked-reduction GEMM -------------------------------
// stats[b][m*6+n]:
//   rows m=0..7 : instance k=m+1 -> n: 0=count 1=Sum x 2=Sum y 3=Sum sig 4=Sum sig^2
//   row  m=8    : background     -> n=5: Sum seed^2
__global__ void k_stats(const float* __restrict__ pred,
                        const int*   __restrict__ inst,
                        const int*   __restrict__ lab,
                        float*       __restrict__ stats) {
  const int b    = blockIdx.x >> 5;     // 32 workgroups per image
  const int wg   = blockIdx.x & 31;
  const int wv   = threadIdx.x >> 5;    // wave in workgroup (8 waves)
  const int lane = threadIdx.x & 31;
  const int waveId = wg * 8 + wv;       // 0..255 per image

  const float* p2 = pred + ((size_t)b * 4 + 2) * HWPX;  // sigma
  const float* p3 = pred + ((size_t)b * 4 + 3) * HWPX;  // seed logit
  const int*   ib = inst + (size_t)b * HWPX;
  const int*   lb = lab  + (size_t)b * HWPX;

  __shared__ unsigned  ids[8][32];
  __shared__ _Float16  hv[8][6][32];

  hv[wv][0][lane] = (_Float16)1.0f;      // loop-invariant "ones" column

  v8f c = {};                            // f32 accumulator (C/D matrix)

  for (int ch = 0; ch < 64; ++ch) {
    const int p = (waveId * 64 + ch) * 32 + lane;
    __builtin_prefetch(p2 + p + 2048, 0, 1);   // global_prefetch_b8

    const float sig  = p2[p];
    const float seed = fast_sigmoid(p3[p]);
    const int   iv   = ib[p];
    const int   lv   = lb[p];
    const int   w    = p & (IMG_W - 1);
    const int   h    = p >> 10;
    const float x    = (float)w * (2.0f / 2047.0f);
    const float y    = (float)h * (1.0f / 1023.0f);

    unsigned idp = ((iv >= 1 && iv <= KMAX) ? (unsigned)(iv - 1) : 0xFFu)
                 | ((lv == 0) ? 0x100u : 0u);
    ids[wv][lane]   = idp;
    hv[wv][1][lane] = (_Float16)x;
    hv[wv][2][lane] = (_Float16)y;
    hv[wv][3][lane] = (_Float16)sig;
    hv[wv][4][lane] = (_Float16)(sig * sig);
    hv[wv][5][lane] = (_Float16)(seed * seed);
    __syncthreads();

    // A fragment (16x32 f16): lane L<16 -> M=L, K in {0..7,16..23};
    //                         lane L>=16 -> M=L-16, K in {8..15,24..31}
    v16h a, bf;
    const int M   = lane & 15;
    const int kbA = (lane < 16) ? 0 : 8;
    #pragma unroll
    for (int hh = 0; hh < 16; ++hh) {
      const int K = (hh & 7) + kbA + ((hh < 8) ? 0 : 16);
      const unsigned id = ids[wv][K];
      float av = 0.0f;
      if (M < 8)        av = ((id & 0xFFu) == (unsigned)M) ? 1.0f : 0.0f;
      else if (M == 8)  av = (id & 0x100u) ? 1.0f : 0.0f;
      a[hh] = (_Float16)av;
    }
    // B fragment (32x16 f16): lane L -> N=L&15, K same as A fragment
    const int N   = lane & 15;
    const int kbB = (lane < 16) ? 0 : 8;
    #pragma unroll
    for (int hh = 0; hh < 16; ++hh) {
      const int K = (hh & 7) + kbB + ((hh < 8) ? 0 : 16);
      bf[hh] = (N < 6) ? hv[wv][N][K] : (_Float16)0.0f;
    }

    c = __builtin_amdgcn_wmma_f32_16x16x32_f16(false, a, false, bf,
                                               (short)0, c, false, false);
    __syncthreads();
  }

  // C/D layout: vgpr j, lanes 0-15 -> M=j, lanes 16-31 -> M=j+8, N=lane&15
  #pragma unroll
  for (int j = 0; j < 8; ++j) {
    const int m = j + ((lane >= 16) ? 8 : 0);
    const int n = lane & 15;
    if (m < 9 && n < 6)
      atomicAdd(&stats[(size_t)b * 64 + m * 6 + n], c[j]);
  }
}

// ---------------- finalize per-instance parameters -------------------------
// params[(b*8+k)*8]: 0=cx 1=cy 2=s_exp 3=present 4=var 5=objf 6=counts
__global__ void k_finalize(const float* __restrict__ stats,
                           float* __restrict__ params,
                           float* __restrict__ lossAcc) {
  const int t = threadIdx.x;             // 0..31
  const int b = t >> 3, k = t & 7;
  const float* S = stats + (size_t)b * 64 + (size_t)k * 6;
  const float counts = S[0], sx = S[1], sy = S[2], ss = S[3], ss2 = S[4];
  const float cnt  = fmaxf(counts, 1.0f);
  const float cx   = sx / cnt, cy = sy / cnt, sm = ss / cnt;
  const float var  = fmaxf(ss2 / cnt - sm * sm, 0.0f);
  const float sexp = expf(10.0f * sm);
  const float present = (counts > 0.0f) ? 1.0f : 0.0f;

  float* pp = params + (size_t)t * 8;
  pp[0] = cx; pp[1] = cy; pp[2] = sexp; pp[3] = present; pp[4] = var; pp[6] = counts;

  __shared__ float pr[32], va[32];
  pr[t] = present; va[t] = present * var;
  __syncthreads();

  if (k == 0) {
    float ps = 0.0f, vs = 0.0f;
    for (int i = 0; i < 8; ++i) { ps += pr[b * 8 + i]; vs += va[b * 8 + i]; }
    const float objf = fmaxf(ps, 1.0f);
    for (int i = 0; i < 8; ++i) params[((size_t)b * 8 + i) * 8 + 5] = objf;
    const float seedbg = stats[(size_t)b * 64 + 8 * 6 + 5];
    lossAcc[b] += W_VAR * vs / objf + W_SEED * seedbg / (float)HWPX;
  }
}

// ---------------- key generation + seed_fg ---------------------------------
__global__ void k_genkeys(const float* __restrict__ pred,
                          const int*   __restrict__ inst,
                          const float* __restrict__ params,
                          unsigned*    __restrict__ keys,
                          float*       __restrict__ lossAcc,
                          int b, int k0) {
  const int sl  = blockIdx.x >> 11;      // 2048 blocks per segment
  const int blk = blockIdx.x & 2047;
  const int t   = threadIdx.x;
  const int k   = k0 + sl;

  const float* pp = params + ((size_t)b * 8 + k) * 8;
  const float cx = pp[0], cy = pp[1], sexp = pp[2], present = pp[3];

  const int p  = blk * 256 + t;
  const float p0 = pred[((size_t)b * 4 + 0) * HWPX + p];
  const float p1 = pred[((size_t)b * 4 + 1) * HWPX + p];
  const float p3 = pred[((size_t)b * 4 + 3) * HWPX + p];
  const int   iv = inst[(size_t)b * HWPX + p];

  const int   w = p & (IMG_W - 1);
  const int   h = p >> 10;
  const float ex = tanhf(p0) + (float)w * (2.0f / 2047.0f);
  const float ey = tanhf(p1) + (float)h * (1.0f / 1023.0f);
  const float dx = ex - cx, dy = ey - cy;
  const float dist = expf(-sexp * (dx * dx + dy * dy));

  const unsigned m = (iv == k + 1) ? 1u : 0u;
  const float err = m ? (2.0f - 2.0f * dist) : (2.0f * dist);   // err >= 0
  unsigned key = (~__float_as_uint(err)) & 0x7fffffffu;         // descending order
  key = (key & ~1u) | m;                                        // gt bit in LSB
  keys[(size_t)sl * HWPX + p] = key;

  // seed_fg contribution: (seed - dist)^2 on foreground pixels of k
  const float seed = fast_sigmoid(p3);
  const float d    = seed - dist;
  __shared__ float red[256];
  const float tot = blockReduce256(m ? d * d : 0.0f, red, t);
  if (t == 0 && present > 0.5f)
    atomicAdd(&lossAcc[b], tot * (W_SEED * FG_W / (float)HWPX));
}

// ---------------- radix sort: histogram / scan / scatter -------------------
__global__ void k_hist(const unsigned* __restrict__ keys,
                       unsigned* __restrict__ hist, int shift) {
  const int sl = blockIdx.x >> 7, blk = blockIdx.x & 127, t = threadIdx.x;
  __shared__ unsigned cnt[256];
  cnt[t] = 0;
  __syncthreads();
  const unsigned* kk = keys + (size_t)sl * HWPX + (size_t)blk * TILE;
  #pragma unroll
  for (int i = 0; i < 16; ++i) {
    const unsigned d = (kk[i * 256 + t] >> shift) & 255u;
    atomicAdd(&cnt[d], 1u);
  }
  __syncthreads();
  hist[((size_t)sl * 256 + t) * NBLK + blk] = cnt[t];
}

__global__ void k_scanhist(unsigned* __restrict__ hist) {
  const int sl = blockIdx.x, d = threadIdx.x;
  unsigned* row = hist + ((size_t)sl * 256 + d) * NBLK;
  unsigned sum = 0;
  for (int i = 0; i < NBLK; ++i) { const unsigned v = row[i]; row[i] = sum; sum += v; }
  __shared__ unsigned tot[256], basearr[256];
  tot[d] = sum;
  __syncthreads();
  if (d == 0) {
    unsigned r = 0;
    for (int i = 0; i < 256; ++i) { basearr[i] = r; r += tot[i]; }
  }
  __syncthreads();
  const unsigned bse = basearr[d];
  for (int i = 0; i < NBLK; ++i) row[i] += bse;
}

__global__ void k_scatter(const unsigned* __restrict__ in,
                          unsigned* __restrict__ out,
                          const unsigned* __restrict__ hist, int shift) {
  const int sl = blockIdx.x >> 7, blk = blockIdx.x & 127, t = threadIdx.x;
  __shared__ __align__(16) unsigned bufA[TILE];
  __shared__ __align__(16) unsigned bufB[TILE];
  __shared__ unsigned s[8];
  __shared__ unsigned gBase[256];
  __shared__ unsigned lStart[256];
  __shared__ unsigned totOnes;

  const unsigned* kk = in + (size_t)sl * HWPX + (size_t)blk * TILE;

  // Async tile load: global -> LDS directly (no VGPR staging), ASYNCcnt.
  // LDS byte address = low 32 bits of the generic LDS pointer.
  {
    const unsigned ldsBase = (unsigned)(size_t)(&bufA[0]);
    const unsigned long long gAddr0 = (unsigned long long)(size_t)kk;
    #pragma unroll
    for (int i = 0; i < 4; ++i) {
      const unsigned           lo = ldsBase + (unsigned)((i * 256 + t) * 16);
      const unsigned long long ga = gAddr0 + (unsigned long long)((i * 256 + t) * 16);
      asm volatile("global_load_async_to_lds_b128 %0, %1, off"
                   :: "v"(lo), "v"(ga) : "memory");
    }
    asm volatile("s_wait_asynccnt 0x0" ::: "memory");
  }
  unsigned* src = bufA;
  unsigned* dst = bufB;
  __syncthreads();

  // stable local sort of the 4096-key tile by the current 8-bit digit
  for (int bit = 0; bit < 8; ++bit) {
    const int bp = shift + bit;
    unsigned loc[16]; int pre[16]; int ones = 0;
    #pragma unroll
    for (int j = 0; j < 16; ++j) {
      loc[j] = src[t * 16 + j];
      pre[j] = ones;
      ones += (int)((loc[j] >> bp) & 1u);
    }
    const unsigned incl = scanBlockIncl((unsigned)ones, s, t);
    const unsigned excl = incl - (unsigned)ones;
    if (t == 255) totOnes = incl;
    __syncthreads();
    const unsigned zt = (unsigned)TILE - totOnes;
    #pragma unroll
    for (int j = 0; j < 16; ++j) {
      const unsigned bb  = (loc[j] >> bp) & 1u;
      const unsigned idx = (unsigned)(t * 16 + j);
      const unsigned ob  = excl + (unsigned)pre[j];
      const unsigned pos = bb ? (zt + ob) : (idx - ob);
      dst[pos] = loc[j];
    }
    __syncthreads();
    unsigned* tmp = src; src = dst; dst = tmp;
  }

  gBase[t] = hist[((size_t)sl * 256 + t) * NBLK + blk];
  #pragma unroll
  for (int j = 0; j < 16; ++j) {
    const int idx = t * 16 + j;
    const unsigned d  = (src[idx] >> shift) & 255u;
    const unsigned dp = (idx > 0) ? ((src[idx - 1] >> shift) & 255u) : 0xFFFFu;
    if (idx == 0 || d != dp) lStart[d] = (unsigned)idx;
  }
  __syncthreads();
  #pragma unroll
  for (int j = 0; j < 16; ++j) {
    const int idx = t * 16 + j;
    const unsigned key = src[idx];
    const unsigned d   = (key >> shift) & 255u;
    const unsigned pos = gBase[d] + (unsigned)idx - lStart[d];
    out[(size_t)sl * HWPX + pos] = key;
  }
}

// ---------------- segmented scan of gt bits --------------------------------
__global__ void k_gtsum(const unsigned* __restrict__ keys,
                        unsigned* __restrict__ gtBlk) {
  const int sl = blockIdx.x >> 7, blk = blockIdx.x & 127, t = threadIdx.x;
  const unsigned* kk = keys + (size_t)sl * HWPX + (size_t)blk * TILE;
  unsigned sum = 0;
  #pragma unroll
  for (int i = 0; i < 16; ++i) sum += kk[i * 256 + t] & 1u;
  __shared__ unsigned red[256];
  red[t] = sum;
  __syncthreads();
  for (int off = 128; off > 0; off >>= 1) {
    if (t < off) red[t] += red[t + off];
    __syncthreads();
  }
  if (t == 0) gtBlk[sl * NBLK + blk] = red[0];
}

__global__ void k_gtscan(unsigned* __restrict__ gtBlk) {
  const int sl = blockIdx.x;
  if (threadIdx.x == 0) {
    unsigned r = 0;
    for (int i = 0; i < NBLK; ++i) {
      const unsigned v = gtBlk[sl * NBLK + i];
      gtBlk[sl * NBLK + i] = r;
      r += v;
    }
  }
}

// ---------------- Lovasz hinge reduction -----------------------------------
// loss_k = sum_i J_i * (e_i - e_{i+1}),  e_N = 0,
// J_i = 1 - (P - cum_i) / (P + (i+1) - cum_i), cum_i = inclusive gt cumsum
__global__ void k_lovasz(const unsigned* __restrict__ keys,
                         const unsigned* __restrict__ gtBlk,
                         const float* __restrict__ params,
                         float* __restrict__ lossAcc,
                         int b, int k0) {
  const int sl = blockIdx.x >> 7, blk = blockIdx.x & 127, t = threadIdx.x;
  const int k  = k0 + sl;
  const float* pp = params + ((size_t)b * 8 + k) * 8;
  const float present = pp[3], objf = pp[5], P = pp[6];

  const unsigned* kk = keys + (size_t)sl * HWPX;
  const int base = blk * TILE;

  unsigned loc[16]; int pre[16]; int ones = 0;
  #pragma unroll
  for (int j = 0; j < 16; ++j) {
    loc[j] = kk[base + t * 16 + j];
    pre[j] = ones;
    ones += (int)(loc[j] & 1u);
  }
  __shared__ unsigned s[8];
  const unsigned incl = scanBlockIncl((unsigned)ones, s, t);
  const unsigned excl = incl - (unsigned)ones;
  const unsigned off0 = gtBlk[sl * NBLK + blk];

  float acc = 0.0f;
  #pragma unroll
  for (int j = 0; j < 16; ++j) {
    const int g = base + t * 16 + j;
    const float cum = (float)(off0 + excl + (unsigned)pre[j] + (loc[j] & 1u));
    const float e   = __uint_as_float((~loc[j]) & 0x7fffffffu);
    float en;
    if (j < 15) en = __uint_as_float((~loc[j + 1]) & 0x7fffffffu);
    else        en = (g + 1 < HWPX) ? __uint_as_float((~kk[g + 1]) & 0x7fffffffu)
                                    : 0.0f;
    const float gp1 = (float)(g + 1);
    const float J = 1.0f - (P - cum) / (P + gp1 - cum);
    acc += J * (e - en);
  }
  __shared__ float red[256];
  const float tot = blockReduce256(acc, red, t);
  if (t == 0 && present > 0.5f)
    atomicAdd(&lossAcc[b], tot * (W_INST / objf));
}

// ---------------- final mean -----------------------------------------------
__global__ void k_final(const float* __restrict__ lossAcc, float* __restrict__ out) {
  if (threadIdx.x == 0)
    out[0] = 0.25f * (lossAcc[0] + lossAcc[1] + lossAcc[2] + lossAcc[3]);
}

// ---------------------------------------------------------------------------
extern "C" void kernel_launch(void* const* d_in, const int* in_sizes, int n_in,
                              void* d_out, int out_size, void* d_ws, size_t ws_size,
                              hipStream_t stream) {
  const float* pred = (const float*)d_in[0];
  const int*   inst = (const int*)d_in[1];
  const int*   lab  = (const int*)d_in[2];
  float*       out  = (float*)d_out;
  char*        ws   = (char*)d_ws;

  // choose segments-per-batch (8,4,2,1) by scratch size
  int nseg = 8;
  while (nseg > 1) {
    const size_t need = 4096
                      + (size_t)nseg * HWPX * 4 * 2          // key ping-pong
                      + (size_t)nseg * 256 * NBLK * 4        // histograms
                      + (size_t)nseg * NBLK * 4;             // gt block sums
    if (need <= ws_size) break;
    nseg >>= 1;
  }

  float*    hdr     = (float*)ws;
  float*    stats   = hdr;          // 256 floats
  float*    params  = hdr + 256;    // 256 floats
  float*    lossAcc = hdr + 512;    // 4 floats
  unsigned* keysA   = (unsigned*)(ws + 4096);
  unsigned* keysB   = keysA + (size_t)nseg * HWPX;
  unsigned* hist    = keysB + (size_t)nseg * HWPX;
  unsigned* gtBlk   = hist  + (size_t)nseg * 256 * NBLK;

  k_zero<<<dim3(1), dim3(256), 0, stream>>>(hdr);
  k_stats<<<dim3(128), dim3(256), 0, stream>>>(pred, inst, lab, stats);
  k_finalize<<<dim3(1), dim3(32), 0, stream>>>(stats, params, lossAcc);

  const int shifts[4] = {0, 8, 16, 24};
  for (int b = 0; b < 4; ++b) {
    for (int k0 = 0; k0 < KMAX; k0 += nseg) {
      k_genkeys<<<dim3(nseg * 2048), dim3(256), 0, stream>>>(
          pred, inst, params, keysA, lossAcc, b, k0);

      unsigned* src = keysA;
      unsigned* dst = keysB;
      for (int pass = 0; pass < 4; ++pass) {
        k_hist<<<dim3(nseg * NBLK), dim3(256), 0, stream>>>(src, hist, shifts[pass]);
        k_scanhist<<<dim3(nseg), dim3(256), 0, stream>>>(hist);
        k_scatter<<<dim3(nseg * NBLK), dim3(256), 0, stream>>>(src, dst, hist,
                                                               shifts[pass]);
        unsigned* tmp = src; src = dst; dst = tmp;
      }
      // after 4 passes sorted keys are back in keysA (== src)
      k_gtsum<<<dim3(nseg * NBLK), dim3(256), 0, stream>>>(src, gtBlk);
      k_gtscan<<<dim3(nseg), dim3(32), 0, stream>>>(gtBlk);
      k_lovasz<<<dim3(nseg * NBLK), dim3(256), 0, stream>>>(src, gtBlk, params,
                                                            lossAcc, b, k0);
    }
  }
  k_final<<<dim3(1), dim3(32), 0, stream>>>(lossAcc, out);
}
